// CustomLoRAAttnProcessor_89859305767308
// MI455X (gfx1250) — matrix operand order repaired and checked by reading, not compiled, not run on hardware
//
#include <hip/hip_runtime.h>

typedef __attribute__((ext_vector_type(16))) _Float16 v16h;
typedef __attribute__((ext_vector_type(8)))  _Float16 v8h;
typedef __attribute__((ext_vector_type(8)))  float    v8f;

#define D_MODEL 1280
#define BATCH   8
#define SEQ     1024
#define HEADS   20
#define HDIM    64
#define NROWS   (BATCH * SEQ)      /* 8192 */
#define RANK    4
#define LORA_SCALE 0.25f
#define ATT_SCALE  0.125f          /* 1/sqrt(64) */

union V16HU { v16h v; v8h h[2]; };

// ---- WMMA fragment loaders (contiguous k order used by this file, wave32) -------------------
// A-matrix 16x32 f16: lane<16 -> row=lane, elems 0..15 = K0..15
//                     lane>=16 -> row=lane-16, elems 0..15 = K16..31
__device__ __forceinline__ v16h load_a_frag(const _Float16* __restrict__ p, int ld,
                                            int row0, int k0) {
    int lane   = threadIdx.x & 31;
    int half16 = lane >> 4;
    const _Float16* base = p + (size_t)(row0 + (lane & 15)) * ld + (k0 + (half16 << 4));
    V16HU u;
    u.h[0] = *(const v8h*)(base);
    u.h[1] = *(const v8h*)(base + 8);
    return u.v;
}

// B-matrix 32x16 f16: lane holds column n = lane&15; elems j -> K = k0 + 16*(lane>=16) + j
// Reads storage laid out [n][k] (i.e. a transposed operand) as one contiguous 32B run.
__device__ __forceinline__ v16h load_b_frag(const _Float16* __restrict__ p, int ld,
                                            int n0, int k0) {
    int lane = threadIdx.x & 31;
    const _Float16* base = p + (size_t)(n0 + (lane & 15)) * ld + k0 + ((lane >> 4) << 4);
    return *(const v16h*)base;
}

__device__ __forceinline__ v8f wmma16(v16h a, v16h b, v8f c) {
    return __builtin_amdgcn_wmma_f32_16x16x32_f16(false, a, false, b, (short)0, c,
                                                  false, false);
}

// ---- small prep kernels -----------------------------------------------------
__global__ void f32_to_f16_kernel(const float* __restrict__ in,
                                  _Float16* __restrict__ out, int n) {
    int i = blockIdx.x * blockDim.x + threadIdx.x;
    if (i < n) out[i] = (_Float16)in[i];
}

// W_eff = W + scale * B(D,R) * A(R,D), emitted as f16
__global__ void fold_weight_kernel(const float* __restrict__ W,
                                   const float* __restrict__ A,
                                   const float* __restrict__ Bm,
                                   _Float16* __restrict__ Wh) {
    int idx = blockIdx.x * blockDim.x + threadIdx.x;
    if (idx >= D_MODEL * D_MODEL) return;
    int dout = idx / D_MODEL, din = idx % D_MODEL;
    float acc = W[idx];
#pragma unroll
    for (int r = 0; r < RANK; ++r)
        acc += LORA_SCALE * Bm[dout * RANK + r] * A[r * D_MODEL + din];
    Wh[idx] = (_Float16)acc;
}

// V [8192 x 1280] -> Vt [B*H][HD][S]  (so V fragments become contiguous B-loads)
__global__ void transpose_v_kernel(const _Float16* __restrict__ Vh,
                                   _Float16* __restrict__ Vt) {
    int idx = blockIdx.x * blockDim.x + threadIdx.x;
    if (idx >= NROWS * D_MODEL) return;
    int row = idx / D_MODEL;           // b*1024 + s
    int col = idx % D_MODEL;           // h*64 + hd
    int b = row >> 10, s = row & 1023;
    int h = col >> 6,  hd = col & 63;
    Vt[(((b * HEADS + h) * HDIM + hd) << 10) + s] = Vh[idx];
}

// ---- GEMM: Out[M,N] = X[M,K] @ W[N,K]^T (+bias). One wave = 64x32 C tile. ----
// Rotated prefetch sized to fit the VGPR budget (no spills):
//  - each A fragment is fetched one WMMA-pair ahead of its use
//  - the two shared B fragments are fetched a full k-step ahead
// Peak live set ~122 VGPRs: 64 acc + 32 B (cur+next) + 16 A (cur+next) + addr.
__global__ void gemm_wmma_kernel(const _Float16* __restrict__ X,
                                 const _Float16* __restrict__ W,
                                 _Float16* __restrict__ outH,
                                 float* __restrict__ outF,
                                 const float* __restrict__ bias,
                                 int M, int N, int K) {
    int wid    = blockIdx.x * (blockDim.x >> 5) + (threadIdx.x >> 5);
    int tilesN = N >> 5;
    int tm = (wid / tilesN) << 6;      // 64 rows per wave
    int tn = (wid % tilesN) << 5;      // 32 cols per wave
    if (tm >= M) return;
    int lane = threadIdx.x & 31;
    int half16 = lane >> 4;

    v8f acc[4][2];
#pragma unroll
    for (int r = 0; r < 4; ++r) { acc[r][0] = (v8f){}; acc[r][1] = (v8f){}; }

    // prologue
    v16h wb0 = load_b_frag(W, K, tn,      0);
    v16h wb1 = load_b_frag(W, K, tn + 16, 0);
    v16h xa  = load_a_frag(X, K, tm,      0);

    for (int k0 = 0; k0 < K; k0 += 32) {
        int kn = (k0 + 32 < K) ? k0 + 32 : 0;   // clamped next step (last reload harmless)
        // r = 0 : prefetch A(r=1), consume A(r=0)
        v16h xa1 = load_a_frag(X, K, tm + 16, k0);
        acc[0][0] = wmma16(xa, wb0, acc[0][0]);
        acc[0][1] = wmma16(xa, wb1, acc[0][1]);
        // r = 1 : prefetch A(r=2)
        v16h xa2 = load_a_frag(X, K, tm + 32, k0);
        acc[1][0] = wmma16(xa1, wb0, acc[1][0]);
        acc[1][1] = wmma16(xa1, wb1, acc[1][1]);
        // r = 2 : prefetch A(r=3) and next-step B0
        v16h xa3  = load_a_frag(X, K, tm + 48, k0);
        v16h nwb0 = load_b_frag(W, K, tn, kn);
        acc[2][0] = wmma16(xa2, wb0, acc[2][0]);
        acc[2][1] = wmma16(xa2, wb1, acc[2][1]);
        // r = 3 : prefetch next-step B1 and A(r=0)
        v16h nwb1 = load_b_frag(W, K, tn + 16, kn);
        v16h nxa  = load_a_frag(X, K, tm, kn);
        acc[3][0] = wmma16(xa3, wb0, acc[3][0]);
        acc[3][1] = wmma16(xa3, wb1, acc[3][1]);
        wb0 = nwb0; wb1 = nwb1; xa = nxa;
    }

    int cn = tn + (lane & 15);
    float b0 = 0.f, b1 = 0.f;
    if (bias) { b0 = bias[cn]; b1 = bias[cn + 16]; }
#pragma unroll
    for (int r = 0; r < 4; ++r) {
#pragma unroll
        for (int v = 0; v < 8; ++v) {
            int rm = tm + 16 * r + v + (half16 << 3);
            if (outH) {
                outH[(size_t)rm * N + cn]      = (_Float16)acc[r][0][v];
                outH[(size_t)rm * N + cn + 16] = (_Float16)acc[r][1][v];
            } else {
                outF[(size_t)rm * N + cn]      = acc[r][0][v] + b0;
                outF[(size_t)rm * N + cn + 16] = acc[r][1][v] + b1;
            }
        }
    }
}

// ---- Flash attention: one wave per (b,h,q-tile of 16 rows) ------------------
__global__ void flash_attn_kernel(const _Float16* __restrict__ Qh,
                                  const _Float16* __restrict__ Kh,
                                  const _Float16* __restrict__ Vt,
                                  _Float16* __restrict__ Ctx) {
    __shared__ _Float16 sP[4][16 * 32];   // per-wave P staging (lane=col -> lane=row)
    int wib   = threadIdx.x >> 5;
    int wid   = blockIdx.x * 4 + wib;     // [0, 160*64)
    int qtile = wid & 63;
    int bh    = wid >> 6;                 // b*20 + h
    int b = bh / HEADS, h = bh % HEADS;
    int lane = threadIdx.x & 31;
    int half16 = lane >> 4;
    const int ld = D_MODEL;

    const _Float16* Qbase = Qh + (size_t)b * SEQ * ld + h * HDIM;
    const _Float16* Kbase = Kh + (size_t)b * SEQ * ld + h * HDIM;
    const _Float16* Vbase = Vt + (size_t)bh * HDIM * SEQ;
    int qrow = qtile * 16;

    v16h qa0 = load_a_frag(Qbase, ld, qrow, 0);
    v16h qa1 = load_a_frag(Qbase, ld, qrow, 32);

    v8f o0 = {}, o1 = {}, o2 = {}, o3 = {};
    float mrow[8], lrow[8];
#pragma unroll
    for (int v = 0; v < 8; ++v) { mrow[v] = -3.0e38f; lrow[v] = 0.f; }

    _Float16* myP = sP[wib];

    for (int kt = 0; kt < SEQ / 32; ++kt) {
        int kc = kt * 32;
        // scores: 16 q-rows x 32 key-cols (two 16x16 f32 acc tiles), K = HD = 64
        v8f s0 = {}, s1 = {};
        s0 = wmma16(qa0, load_b_frag(Kbase, ld, kc,      0),  s0);
        s0 = wmma16(qa1, load_b_frag(Kbase, ld, kc,      32), s0);
        s1 = wmma16(qa0, load_b_frag(Kbase, ld, kc + 16, 0),  s1);
        s1 = wmma16(qa1, load_b_frag(Kbase, ld, kc + 16, 32), s1);

        // issue V loads early: they overlap the softmax VALU/shuffle work below
        v16h vb0 = load_b_frag(Vbase, SEQ, 0,  kc);
        v16h vb1 = load_b_frag(Vbase, SEQ, 16, kc);
        v16h vb2 = load_b_frag(Vbase, SEQ, 32, kc);
        v16h vb3 = load_b_frag(Vbase, SEQ, 48, kc);

        float rs[8];
#pragma unroll
        for (int v = 0; v < 8; ++v) {
            float a = s0[v] * ATT_SCALE;
            float c = s1[v] * ATT_SCALE;
            // row max over 32 cols: reduce across the 16 lanes of this half-wave
            float mx = fmaxf(a, c);
            mx = fmaxf(mx, __shfl_xor(mx, 1));
            mx = fmaxf(mx, __shfl_xor(mx, 2));
            mx = fmaxf(mx, __shfl_xor(mx, 4));
            mx = fmaxf(mx, __shfl_xor(mx, 8));
            float mnew = fmaxf(mrow[v], mx);
            float r    = __expf(mrow[v] - mnew);
            mrow[v] = mnew;
            rs[v]   = r;
            float p0 = __expf(a - mnew);
            float p1 = __expf(c - mnew);
            float t  = p0 + p1;
            t += __shfl_xor(t, 1);
            t += __shfl_xor(t, 2);
            t += __shfl_xor(t, 4);
            t += __shfl_xor(t, 8);
            lrow[v] = lrow[v] * r + t;
            // stage P (f16) row-major [16][32]; this lane owns col lane&15 (+16)
            int roff = (v + (half16 << 3)) * 32;
            myP[roff + (lane & 15)]      = (_Float16)p0;
            myP[roff + (lane & 15) + 16] = (_Float16)p1;
        }
        // rescale running O by exp(m_old - m_new); row layout matches acc layout
#pragma unroll
        for (int v = 0; v < 8; ++v) {
            o0[v] *= rs[v]; o1[v] *= rs[v]; o2[v] *= rs[v]; o3[v] *= rs[v];
        }
        __syncthreads();   // uniform across block; orders LDS store->load
        v16h pa = load_a_frag(myP, 32, 0, 0);          // P as 16x32 A-fragment
        o0 = wmma16(pa, vb0, o0);
        o1 = wmma16(pa, vb1, o1);
        o2 = wmma16(pa, vb2, o2);
        o3 = wmma16(pa, vb3, o3);
        __syncthreads();
    }

    // normalize and emit ctx (f16) back in [B*S, D] layout
#pragma unroll
    for (int v = 0; v < 8; ++v) {
        float inv = 1.0f / lrow[v];
        int r = qrow + v + (half16 << 3);
        size_t rowoff = ((size_t)b * SEQ + r) * ld + h * HDIM + (lane & 15);
        Ctx[rowoff + 0]  = (_Float16)(o0[v] * inv);
        Ctx[rowoff + 16] = (_Float16)(o1[v] * inv);
        Ctx[rowoff + 32] = (_Float16)(o2[v] * inv);
        Ctx[rowoff + 48] = (_Float16)(o3[v] * inv);
    }
}

// ---- host-side orchestration ------------------------------------------------
extern "C" void kernel_launch(void* const* d_in, const int* in_sizes, int n_in,
                              void* d_out, int out_size, void* d_ws, size_t ws_size,
                              hipStream_t stream) {
    (void)in_sizes; (void)n_in; (void)out_size; (void)ws_size;
    const float* x  = (const float*)d_in[0];
    const float* Wq = (const float*)d_in[1];
    const float* Wk = (const float*)d_in[2];
    const float* Wv = (const float*)d_in[3];
    const float* Wo = (const float*)d_in[4];
    const float* bo = (const float*)d_in[5];
    const float* Aq = (const float*)d_in[6];
    const float* Bq = (const float*)d_in[7];
    const float* Ak = (const float*)d_in[8];
    const float* Bk = (const float*)d_in[9];
    const float* Av = (const float*)d_in[10];
    const float* Bv = (const float*)d_in[11];
    const float* Ao = (const float*)d_in[12];
    const float* Bo = (const float*)d_in[13];
    float* out = (float*)d_out;

    char* ws = (char*)d_ws;
    size_t off = 0;
    auto carve = [&](size_t bytes) -> char* {
        char* p = ws + off;
        off += (bytes + 255) & ~(size_t)255;
        return p;
    };
    const size_t actBytes = (size_t)NROWS * D_MODEL * sizeof(_Float16);
    const size_t wBytes   = (size_t)D_MODEL * D_MODEL * sizeof(_Float16);
    _Float16* Xh  = (_Float16*)carve(actBytes);   // later reused as Ctx
    _Float16* Wqh = (_Float16*)carve(wBytes);
    _Float16* Wkh = (_Float16*)carve(wBytes);
    _Float16* Wvh = (_Float16*)carve(wBytes);
    _Float16* Woh = (_Float16*)carve(wBytes);
    _Float16* Qh  = (_Float16*)carve(actBytes);
    _Float16* Kh  = (_Float16*)carve(actBytes);
    _Float16* Vh  = (_Float16*)carve(actBytes);
    _Float16* Vt  = (_Float16*)carve(actBytes);
    _Float16* Ctx = Xh;   // Xh dead after the QKV GEMMs

    const int nAct = NROWS * D_MODEL;            // 10,485,760
    const int nW   = D_MODEL * D_MODEL;          // 1,638,400

    f32_to_f16_kernel<<<(nAct + 255) / 256, 256, 0, stream>>>(x, Xh, nAct);

    fold_weight_kernel<<<(nW + 255) / 256, 256, 0, stream>>>(Wq, Aq, Bq, Wqh);
    fold_weight_kernel<<<(nW + 255) / 256, 256, 0, stream>>>(Wk, Ak, Bk, Wkh);
    fold_weight_kernel<<<(nW + 255) / 256, 256, 0, stream>>>(Wv, Av, Bv, Wvh);
    fold_weight_kernel<<<(nW + 255) / 256, 256, 0, stream>>>(Wo, Ao, Bo, Woh);

    // (8192/64)*(1280/32) = 5120 waves, 4 waves/block -> 1280 blocks exactly
    const int gemmBlocks = (NROWS / 64) * (D_MODEL / 32) / 4;
    gemm_wmma_kernel<<<gemmBlocks, 128, 0, stream>>>(Xh, Wqh, Qh, nullptr, nullptr,
                                                     NROWS, D_MODEL, D_MODEL);
    gemm_wmma_kernel<<<gemmBlocks, 128, 0, stream>>>(Xh, Wkh, Kh, nullptr, nullptr,
                                                     NROWS, D_MODEL, D_MODEL);
    gemm_wmma_kernel<<<gemmBlocks, 128, 0, stream>>>(Xh, Wvh, Vh, nullptr, nullptr,
                                                     NROWS, D_MODEL, D_MODEL);

    transpose_v_kernel<<<(nAct + 255) / 256, 256, 0, stream>>>(Vh, Vt);

    // 8*20*64 = 10240 wave-tiles, 4 waves/block -> 2560 blocks exactly
    flash_attn_kernel<<<(BATCH * HEADS * (SEQ / 16)) / 4, 128, 0, stream>>>(Qh, Kh, Vt, Ctx);

    gemm_wmma_kernel<<<gemmBlocks, 128, 0, stream>>>(Ctx, Woh, nullptr, out, bo,
                                                     NROWS, D_MODEL, D_MODEL);
}
